// CellLayer_4183298146536
// MI455X (gfx1250) — matrix operand order repaired and checked by reading, not compiled, not run on hardware
//
#include <hip/hip_runtime.h>
#include <hip/hip_bf16.h>
#include <math.h>

typedef __attribute__((ext_vector_type(16))) __bf16 v16bf;
typedef __attribute__((ext_vector_type(8)))  float  v8f;

union FragBF { v16bf v; unsigned u[8]; };

#define SEQ_LEN 8192
#define NIN     1024
#define NHID    1024
#define N3H     3072

// ---------------------------------------------------------------------------
// bf16 pack helpers (round-to-nearest-even)
// ---------------------------------------------------------------------------
__device__ __forceinline__ unsigned pk2bf(float lo, float hi) {
    unsigned a = __float_as_uint(lo), b = __float_as_uint(hi);
    a = (a + 0x7FFFu + ((a >> 16) & 1u)) >> 16;
    b = (b + 0x7FFFu + ((b >> 16) & 1u)) >> 16;
    return (a & 0xFFFFu) | (b << 16);
}

// ---------------------------------------------------------------------------
// Kernel 1: x_proj[T,3H] = xs[T,K] @ w_ih[3H,K]^T + b  (bf16 WMMA, fp32 acc)
// Block tile 128x128, BK=32, 256 threads (8 waves), wave tile 64x32.
// ---------------------------------------------------------------------------
#define LDT 40   // ushort pitch (80B = 20 banks) to kill bank conflicts

__global__ __launch_bounds__(256, 2)
void xproj_gemm_kernel(const float* __restrict__ A,     // xs   [8192,1024]
                       const float* __restrict__ B,     // w_ih [3072,1024]
                       const float* __restrict__ bias,  // b    [3072]
                       float* __restrict__ C)           // out  [8192,3072]
{
    __shared__ unsigned short As[128 * LDT];
    __shared__ unsigned short Bs[128 * LDT];

    const int tid  = threadIdx.x;
    const int m0   = blockIdx.y * 128;
    const int n0   = blockIdx.x * 128;
    const int w    = tid >> 5;
    const int lane = tid & 31;
    const int m_off = (w & 1) * 64;   // 2 wave rows of 64
    const int n_off = (w >> 1) * 32;  // 4 wave cols of 32
    const int lm = lane & 15;
    const int g  = lane >> 4;

    v8f acc[4][2];
    const v8f z8 = {0.f, 0.f, 0.f, 0.f, 0.f, 0.f, 0.f, 0.f};
#pragma unroll
    for (int fm = 0; fm < 4; ++fm)
#pragma unroll
        for (int fn = 0; fn < 2; ++fn) acc[fm][fn] = z8;

    for (int k0 = 0; k0 < NIN; k0 += 32) {
        __syncthreads();
        // cooperative load+convert: 128 rows x 32 cols each for A and B
#pragma unroll
        for (int i = 0; i < 4; ++i) {
            int q   = tid + i * 256;      // float4 index, 1024 total
            int row = q >> 3;             // 8 float4 per row
            int c4  = q & 7;
            float4 av = *(const float4*)(A + (size_t)(m0 + row) * NIN + k0 + c4 * 4);
            float4 bv = *(const float4*)(B + (size_t)(n0 + row) * NIN + k0 + c4 * 4);
            uint2 ap, bp;
            ap.x = pk2bf(av.x, av.y); ap.y = pk2bf(av.z, av.w);
            bp.x = pk2bf(bv.x, bv.y); bp.y = pk2bf(bv.z, bv.w);
            *(uint2*)&As[row * LDT + c4 * 4] = ap;
            *(uint2*)&Bs[row * LDT + c4 * 4] = bp;
        }
        if (k0 + 32 < NIN) {  // prefetch next K tile -> global_prefetch_b8
            int row = tid >> 3, c4 = tid & 7;
            __builtin_prefetch(A + (size_t)(m0 + row) * NIN + (k0 + 32) + c4 * 4, 0, 1);
            __builtin_prefetch(B + (size_t)(n0 + row) * NIN + (k0 + 32) + c4 * 4, 0, 1);
        }
        __syncthreads();

        // A fragments (16x32, MxK): lane -> M=lm; elem pair p -> K per ISA layout
        FragBF afr[4], bfr[2];
#pragma unroll
        for (int fm = 0; fm < 4; ++fm) {
            int row = m_off + fm * 16 + lm;
#pragma unroll
            for (int p = 0; p < 8; ++p) {
                int K = ((p & 3) << 1) + ((p >> 2) << 4) + (g << 3);
                afr[fm].u[p] = *(const unsigned*)&As[row * LDT + K];
            }
        }
        // B fragments (32x16, KxN): lane -> N=lm; elem pair p -> K per ISA layout (same as A)
#pragma unroll
        for (int fn = 0; fn < 2; ++fn) {
            int col = n_off + fn * 16 + lm;
#pragma unroll
            for (int p = 0; p < 8; ++p) {
                int K = ((p & 3) << 1) + ((p >> 2) << 4) + (g << 3);
                bfr[fn].u[p] = *(const unsigned*)&Bs[col * LDT + K];
            }
        }
#pragma unroll
        for (int fm = 0; fm < 4; ++fm)
#pragma unroll
            for (int fn = 0; fn < 2; ++fn)
                acc[fm][fn] = __builtin_amdgcn_wmma_f32_16x16x32_bf16(
                    false, afr[fm].v, false, bfr[fn].v,
                    (short)0, acc[fm][fn], false, false);
    }

    // epilogue: C/D layout -> lane N=lm (+16 group), vgpr v -> M = v + 8*g
#pragma unroll
    for (int fn = 0; fn < 2; ++fn) {
        int N  = n0 + n_off + fn * 16 + lm;
        float bv = bias[N];
#pragma unroll
        for (int fm = 0; fm < 4; ++fm) {
            int Mb = m0 + m_off + fm * 16;
#pragma unroll
            for (int v = 0; v < 8; ++v) {
                int M = Mb + v + 8 * g;
                C[(size_t)M * N3H + N] = acc[fm][fn][v] + bv;
            }
        }
    }
}

// ---------------------------------------------------------------------------
// Kernel 2: init workspace state (h0 = 0, barrier counters = 0) every launch
// ---------------------------------------------------------------------------
__global__ void init_state_kernel(float* hbufs, unsigned* ctrs) {
    int i = threadIdx.x;
    for (int k = i; k < 2 * NHID; k += 256) hbufs[k] = 0.f;
    if (i < 2) ctrs[i] = 0u;
}

// ---------------------------------------------------------------------------
// Kernel 3: persistent GRU scan. 128 WGs x 192 threads. Each WG owns 8 output
// channels -> 24 rows of w_hh cached in LDS (96KB) for all 8192 steps.
// Device-wide barrier per step via atomic counter + epoch in L2.
// ---------------------------------------------------------------------------
#define NWG 128
#define TPB 192

__global__ __launch_bounds__(TPB, 1)
void gru_scan_kernel(const float* __restrict__ w_hh,   // [3072,1024]
                     const float* __restrict__ bn,     // [1024]
                     const float* __restrict__ xproj,  // [8192,3072]
                     float* __restrict__ out,          // [2,8192,1024]
                     float* hbuf0, float* hbuf1,
                     unsigned* cnt, unsigned* epoch)
{
    extern __shared__ float smem[];
    float* wlds   = smem;               // 24*1024
    float* hlds   = smem + 24 * 1024;   // 1024
    float* rowsum = hlds + 1024;        // 24

    const int tid = threadIdx.x;
    const int wg  = blockIdx.x;

    // Load this WG's 24 rows of w_hh: local row r = g*8 + c  <->  global row g*1024 + wg*8 + c
    for (int idx = tid; idx < 24 * 256; idx += TPB) {   // 6144 float4
        int r  = idx >> 8;
        int c4 = idx & 255;
        int gg = r >> 3, c = r & 7;
        const float4* src = (const float4*)(w_hh + (size_t)(gg * NHID + wg * 8 + c) * NHID) + c4;
        ((float4*)(wlds + r * NHID))[c4] = *src;
    }

    const int rg = tid >> 5;   // 6 row-groups, one wave each; rows 4*rg..4*rg+3
    const int ct = tid & 31;
    unsigned p = 0;

    for (int t = 0; t < SEQ_LEN; ++t) {
        const float* hsrc = p ? hbuf1 : hbuf0;
        float*       hdst = p ? hbuf0 : hbuf1;

        // stage h into LDS
        for (int i = tid; i < NHID; i += TPB) hlds[i] = hsrc[i];
        __syncthreads();

        // matvec: thread handles 4 rows x (32 strided float4 columns)
        float a0 = 0.f, a1 = 0.f, a2 = 0.f, a3 = 0.f;
        const float* w0 = wlds + (rg * 4 + 0) * NHID;
        const float* w1 = wlds + (rg * 4 + 1) * NHID;
        const float* w2 = wlds + (rg * 4 + 2) * NHID;
        const float* w3 = wlds + (rg * 4 + 3) * NHID;
#pragma unroll
        for (int i = 0; i < 8; ++i) {
            int cb = ct * 4 + i * 128;
            float4 hv = *(const float4*)(hlds + cb);
            float4 wv;
            wv = *(const float4*)(w0 + cb); a0 += hv.x * wv.x + hv.y * wv.y + hv.z * wv.z + hv.w * wv.w;
            wv = *(const float4*)(w1 + cb); a1 += hv.x * wv.x + hv.y * wv.y + hv.z * wv.z + hv.w * wv.w;
            wv = *(const float4*)(w2 + cb); a2 += hv.x * wv.x + hv.y * wv.y + hv.z * wv.z + hv.w * wv.w;
            wv = *(const float4*)(w3 + cb); a3 += hv.x * wv.x + hv.y * wv.y + hv.z * wv.z + hv.w * wv.w;
        }
#pragma unroll
        for (int off = 16; off >= 1; off >>= 1) {
            a0 += __shfl_xor(a0, off, 32);
            a1 += __shfl_xor(a1, off, 32);
            a2 += __shfl_xor(a2, off, 32);
            a3 += __shfl_xor(a3, off, 32);
        }
        if (ct == 0) {
            rowsum[rg * 4 + 0] = a0; rowsum[rg * 4 + 1] = a1;
            rowsum[rg * 4 + 2] = a2; rowsum[rg * 4 + 3] = a3;
        }
        __syncthreads();

        // gate math for this WG's 8 channels
        if (tid < 8) {
            int j = wg * 8 + tid;
            float hr = rowsum[tid], hz = rowsum[8 + tid], hg = rowsum[16 + tid];
            const float* xp = xproj + (size_t)t * N3H;
            float ir = xp[j], iz = xp[NHID + j], ig = xp[2 * NHID + j];
            float r  = 1.f / (1.f + __expf(-(ir + hr)));
            float zz = 1.f / (1.f + __expf(-(iz + hz)));
            float gv = tanhf(ig + r * (hg + bn[j]));
            float hn = (1.f - zz) * gv + zz * hlds[j];
            hdst[j] = hn;
            out[(size_t)t * NHID + j] = hn;
            out[(size_t)SEQ_LEN * NHID + (size_t)t * NHID + j] = hn;
        }

        // device-wide barrier (sense via monotonic epoch)
        __threadfence();
        __syncthreads();
        if (tid == 0) {
            unsigned arrived = __hip_atomic_fetch_add(cnt, 1u, __ATOMIC_ACQ_REL,
                                                      __HIP_MEMORY_SCOPE_AGENT);
            unsigned target = (unsigned)(t + 1) * NWG;
            if (arrived == target - 1u) {
                __hip_atomic_store(epoch, (unsigned)(t + 1), __ATOMIC_RELEASE,
                                   __HIP_MEMORY_SCOPE_AGENT);
            } else {
                while (__hip_atomic_load(epoch, __ATOMIC_ACQUIRE,
                                         __HIP_MEMORY_SCOPE_AGENT) < (unsigned)(t + 1)) {
                    __builtin_amdgcn_s_sleep(2);
                }
            }
        }
        __syncthreads();
        __threadfence();
        p ^= 1u;
    }
}

// ---------------------------------------------------------------------------
extern "C" void kernel_launch(void* const* d_in, const int* in_sizes, int n_in,
                              void* d_out, int out_size, void* d_ws, size_t ws_size,
                              hipStream_t stream) {
    const float* xs   = (const float*)d_in[0];
    const float* w_ih = (const float*)d_in[1];
    const float* w_hh = (const float*)d_in[2];
    const float* b    = (const float*)d_in[3];
    const float* bn   = (const float*)d_in[4];
    float* out = (float*)d_out;

    char*  ws     = (char*)d_ws;
    float* xproj  = (float*)ws;                               // 8192*3072 f32 = 96MB
    size_t xbytes = (size_t)SEQ_LEN * N3H * sizeof(float);
    float* hbuf0  = (float*)(ws + xbytes);
    float* hbuf1  = hbuf0 + NHID;
    unsigned* ctrs = (unsigned*)(hbuf1 + NHID);               // [cnt, epoch]

    init_state_kernel<<<1, 256, 0, stream>>>(hbuf0, ctrs);

    dim3 gg(N3H / 128, SEQ_LEN / 128);                        // (24, 64)
    xproj_gemm_kernel<<<gg, 256, 0, stream>>>(xs, w_ih, b, xproj);

    size_t smem = (size_t)(24 * NHID + NHID + 32) * sizeof(float);  // ~100.1 KB
    gru_scan_kernel<<<NWG, TPB, smem, stream>>>(w_hh, bn, xproj, out,
                                                hbuf0, hbuf1, ctrs, ctrs + 1);
}
